// Transcoder_53747220742705
// MI455X (gfx1250) — matrix operand order repaired and checked by reading, not compiled, not run on hardware
//
#include <hip/hip_runtime.h>
#include <hip/hip_bf16.h>

// ---------------------------------------------------------------------------
// TopK-SAE transcoder for MI455X (gfx1250, wave32, WMMA + TDM).
//   prep   : X, W_enc -> bf16 (once); W_dec -> W_decT (once)
//   encode : pre_act = X @ W_enc^T + b_enc   (bf16 WMMA, f32 acc, TDM-staged LDS)
//   topk   : per-row exact top-32 via 8-bit MSB radix select in LDS
//   decode : pred[b,:] = sum_k val_k * W_decT[idx_k,:]
// d_out = [ pred (4096*768 f32) | latents (4096*24576 f32) ]
// d_ws  = [ W_decT f32 | W_enc bf16 | X bf16 | topk_idx | topk_val ]
// ---------------------------------------------------------------------------

#define D_MODEL 768
#define D_SAE   24576
#define KSEL    32
#define BATCH   4096

typedef __attribute__((ext_vector_type(16))) __bf16          v16bf;
typedef __attribute__((ext_vector_type(8)))  float           v8f;
typedef __attribute__((ext_vector_type(8)))  unsigned short  u16x8;
typedef __attribute__((ext_vector_type(4)))  unsigned short  u16x4;
typedef __attribute__((ext_vector_type(4)))  unsigned int    v4u;
typedef __attribute__((ext_vector_type(8)))  int             v8i;
typedef __attribute__((ext_vector_type(4)))  int             v4i;

union FragBF { v16bf bf; u16x8 h[2]; };

__device__ __forceinline__ unsigned short f2bf_rne(float f) {
    unsigned u = __float_as_uint(f);
    return (unsigned short)((u + 0x7FFFu + ((u >> 16) & 1u)) >> 16);
}

__device__ __forceinline__ unsigned fkey(float f) {
    unsigned u = __float_as_uint(f);
    return u ^ ((unsigned)(((int)u) >> 31) | 0x80000000u);
}

// ---------------------------------------------------------------------------
// TDM: DMA a 2-D bf16 tile (tile_d0 cols x tile_d1 rows, row stride stride0
// elements) from global to LDS, padding each 16-DWORD row with 4 DWORDs so the
// LDS image has 40-element (80 B) row pitch (bank-conflict-light, 16B-aligned).
// D# layout per CDNA5 ISA 8.3/8.4.  One issue per wave; tracked by TENSORcnt.
// ---------------------------------------------------------------------------
__device__ __forceinline__ void tdm_load_2d_bf16(unsigned lds_byte_addr,
                                                 const void* gptr,
                                                 unsigned tensor_d0, unsigned tensor_d1,
                                                 unsigned tile_d0, unsigned tile_d1,
                                                 unsigned stride0) {
    unsigned long long ga = (unsigned long long)gptr;
    v4u g0 = { 1u,                                          // count=1, user D#
               lds_byte_addr,                               // lds_addr
               (unsigned)(ga & 0xFFFFFFFFull),              // global_addr[31:0]
               (unsigned)((ga >> 32) & 0x01FFFFFFull)       // global_addr[56:32]
                   | (2u << 30) };                          // type=2 (image)
    v8i g1 = { (int)((1u << 16)                             // data_size = 2B
                   | (1u << 20)                             // pad_enable
                   | (3u << 22)                             // pad_interval: 16 DW
                   | (3u << 25)),                           // pad_amount: 4 DW
               (int)(tensor_d0 << 16),                      // tensor_dim0[15:0]
               (int)((tensor_d0 >> 16) | (tensor_d1 << 16)),
               (int)((tensor_d1 >> 16) | (tile_d0 << 16)),
               (int)(tile_d1),                              // tile_dim1 (tile_dim2=0)
               (int)(stride0),                              // tensor_dim0_stride[31:0]
               0, 0 };
    v4i z4 = {0, 0, 0, 0};
#if defined(__clang_major__) && (__clang_major__ >= 23)
    v8i z8 = {0, 0, 0, 0, 0, 0, 0, 0};
    __builtin_amdgcn_tensor_load_to_lds(g0, g1, z4, z4, z8, 0);
#else
    __builtin_amdgcn_tensor_load_to_lds(g0, g1, z4, z4, 0);
#endif
}

// ---------------------------------------------------------------------------
// Kernel 0: f32 -> bf16 elementwise (RNE), vectorized x4.
// ---------------------------------------------------------------------------
__global__ __launch_bounds__(256) void f32_to_bf16(const float* __restrict__ src,
                                                   unsigned short* __restrict__ dst,
                                                   int n4) {
    const int i = blockIdx.x * 256 + threadIdx.x;
    if (i < n4) {
        float4 v = ((const float4*)src)[i];
        u16x4 h = {f2bf_rne(v.x), f2bf_rne(v.y), f2bf_rne(v.z), f2bf_rne(v.w)};
        ((u16x4*)dst)[i] = h;
    }
}

// ---------------------------------------------------------------------------
// Kernel 1: transpose W_dec [768, 24576] -> W_decT [24576, 768]
// ---------------------------------------------------------------------------
__global__ __launch_bounds__(256) void wdec_transpose(const float* __restrict__ Wd,
                                                      float* __restrict__ WdT) {
    __shared__ float tile[32][33];
    const int tx = threadIdx.x & 31;
    const int ty = threadIdx.x >> 5;
    const int bs = blockIdx.x * 32;
    const int bd = blockIdx.y * 32;
    #pragma unroll
    for (int j = 0; j < 32; j += 8)
        tile[ty + j][tx] = Wd[(size_t)(bd + ty + j) * D_SAE + bs + tx];
    __syncthreads();
    #pragma unroll
    for (int j = 0; j < 32; j += 8)
        WdT[(size_t)(bs + ty + j) * D_MODEL + bd + tx] = tile[tx][ty + j];
}

// ---------------------------------------------------------------------------
// Kernel 2: encode GEMM. Block tile 128x128x32, 8 waves, wave tile 64x32
// (4x2 v_wmma_f32_16x16x32_bf16). Double-buffered LDS staged by the Tensor
// Data Mover: each wave DMAs one 32-row slice (waves 0-3: A, waves 4-7: B),
// then s_wait_tensorcnt(0) + one workgroup barrier per K-step. TDM issue for
// tile k+1 overlaps WMMA on tile k.
// ---------------------------------------------------------------------------
__global__ __launch_bounds__(256) void encode_gemm(const unsigned short* __restrict__ Abf,
                                                   const unsigned short* __restrict__ Wbf,
                                                   const float* __restrict__ bias,
                                                   float* __restrict__ preact) {
    __shared__ unsigned short As[2][128 * 40];
    __shared__ unsigned short Bs[2][128 * 40];

    const int tid  = threadIdx.x;
    const int lane = tid & 31;
    const int wid  = tid >> 5;
    const int wm   = wid >> 2;                 // 0..1
    const int wn   = wid & 3;                  // 0..3
    const int bm   = blockIdx.x * 128;
    const int bn   = blockIdx.y * 128;

    v8f acc[4][2];
    #pragma unroll
    for (int mi = 0; mi < 4; ++mi)
        #pragma unroll
        for (int ni = 0; ni < 2; ++ni)
            acc[mi][ni] = (v8f){0.f, 0.f, 0.f, 0.f, 0.f, 0.f, 0.f, 0.f};

    const int cn = lane & 15;
    float bv[2];
    #pragma unroll
    for (int ni = 0; ni < 2; ++ni)
        bv[ni] = bias[bn + wn * 32 + ni * 16 + cn];

    // Per-wave TDM slice: waves 0-3 -> A rows [32w,32w+32), waves 4-7 -> B.
    const int isB   = wid >> 2 >= 1 ? ((wid >= 4) ? 1 : 0) : 0;
    const int sl    = (wid & 3) * 32;          // row offset inside 128-row tile
    const unsigned short* gsl = (wid >= 4) ? (Wbf + (size_t)(bn + sl) * D_MODEL)
                                           : (Abf + (size_t)(bm + sl) * D_MODEL);
    unsigned ldsSl[2];
    ldsSl[0] = (unsigned)(uintptr_t)((wid >= 4) ? &Bs[0][0] : &As[0][0]) + (unsigned)(sl * 40 * 2);
    ldsSl[1] = (unsigned)(uintptr_t)((wid >= 4) ? &Bs[1][0] : &As[1][0]) + (unsigned)(sl * 40 * 2);
    (void)isB;

    // Prologue: DMA K-tile 0 into buffer 0.
    tdm_load_2d_bf16(ldsSl[0], gsl, D_MODEL, 32, 32, 32, D_MODEL);

    for (int kt = 0; kt < D_MODEL / 32; ++kt) {
        const int cur = kt & 1;
        __builtin_amdgcn_s_wait_tensorcnt(0);  // my slice of buf[cur] landed
        __syncthreads();                       // all slices landed; prev compute done
        if (kt + 1 < D_MODEL / 32)             // overlap: DMA next tile into buf[cur^1]
            tdm_load_2d_bf16(ldsSl[cur ^ 1], gsl + (size_t)(kt + 1) * 32,
                             D_MODEL, 32, 32, 32, D_MODEL);

        // A frag: lanes 0-15 -> K 0..7 & 16..23 ; lanes 16-31 -> K 8..15 & 24..31
        FragBF af[4];
        #pragma unroll
        for (int mi = 0; mi < 4; ++mi) {
            const int r  = wm * 64 + mi * 16 + (lane & 15);
            const int k0 = (lane < 16) ? 0 : 8;
            af[mi].h[0] = *(const u16x8*)(&As[cur][r * 40 + k0]);
            af[mi].h[1] = *(const u16x8*)(&As[cur][r * 40 + k0 + 16]);
        }
        // B frag: lane-per-column; lanes 0-15 -> K 0..7 & 16..23 ; lanes 16-31 -> K 8..15 & 24..31
        FragBF bf[2];
        #pragma unroll
        for (int ni = 0; ni < 2; ++ni) {
            const int c  = wn * 32 + ni * 16 + (lane & 15);
            const int k0 = (lane < 16) ? 0 : 8;
            bf[ni].h[0] = *(const u16x8*)(&Bs[cur][c * 40 + k0]);
            bf[ni].h[1] = *(const u16x8*)(&Bs[cur][c * 40 + k0 + 16]);
        }
        #pragma unroll
        for (int mi = 0; mi < 4; ++mi)
            #pragma unroll
            for (int ni = 0; ni < 2; ++ni)
                acc[mi][ni] = __builtin_amdgcn_wmma_f32_16x16x32_bf16(
                    false, af[mi].bf, false, bf[ni].bf,
                    (short)0, acc[mi][ni], false, false);
    }

    // C/D layout: VGPR r, lanes 0-15: M=r, N=lane ; lanes 16-31: M=r+8, N=lane-16
    const int rlo = (lane >> 4) * 8;
    #pragma unroll
    for (int mi = 0; mi < 4; ++mi) {
        const int rowG0 = bm + wm * 64 + mi * 16 + rlo;
        #pragma unroll
        for (int ni = 0; ni < 2; ++ni) {
            const size_t colG = (size_t)(bn + wn * 32 + ni * 16 + cn);
            #pragma unroll
            for (int r = 0; r < 8; ++r)
                preact[(size_t)(rowG0 + r) * D_SAE + colG] = acc[mi][ni][r] + bv[ni];
        }
    }
}

// ---------------------------------------------------------------------------
// Kernel 3: exact top-32 per row (row cached in 96KB dynamic LDS; 320KB WGP).
// 4x 8-bit MSB radix passes; deterministic index-order tie-break via chunk
// counts + exclusive scan (stable positions, no atomic-order nondeterminism).
// ---------------------------------------------------------------------------
__global__ __launch_bounds__(256) void topk_select(float* __restrict__ latents,
                                                   int* __restrict__ tIdx,
                                                   float* __restrict__ tVal) {
    extern __shared__ float row[];
    __shared__ unsigned hist[256];
    __shared__ unsigned sGt[256], sEq[256];
    __shared__ unsigned sPrefix, sRemaining;

    const int tid = threadIdx.x;
    const size_t b = blockIdx.x;
    float* g = latents + b * D_SAE;

    for (int i = tid; i < D_SAE; i += 256) row[i] = g[i];
    if (tid == 0) { sPrefix = 0u; sRemaining = KSEL; }

    for (int shift = 24; shift >= 0; shift -= 8) {
        hist[tid] = 0u;
        __syncthreads();
        const unsigned pfx = sPrefix;
        for (int i = tid; i < D_SAE; i += 256) {
            const unsigned k = fkey(row[i]);
            if (shift == 24 || (k >> (shift + 8)) == pfx)
                atomicAdd(&hist[(k >> shift) & 255u], 1u);
        }
        __syncthreads();
        if (tid == 0) {
            unsigned rem = sRemaining, cum = 0u;
            for (int d = 255; d >= 0; --d) {
                const unsigned h = hist[d];
                if (cum + h >= rem) {
                    sPrefix = (pfx << 8) | (unsigned)d;
                    sRemaining = rem - cum;
                    break;
                }
                cum += h;
            }
        }
        __syncthreads();
    }
    const unsigned T = sPrefix;
    const unsigned tieNeed = sRemaining;

    const int CH = D_SAE / 256;
    const int base = tid * CH;
    unsigned gtc = 0u, eqc = 0u;
    for (int j = 0; j < CH; ++j) {
        const unsigned k = fkey(row[base + j]);
        gtc += (k > T);
        eqc += (k == T);
    }
    sGt[tid] = gtc; sEq[tid] = eqc;
    __syncthreads();
    if (tid == 0) {
        unsigned ag = 0u, ae = 0u;
        for (int t = 0; t < 256; ++t) {
            const unsigned g1 = sGt[t], e1 = sEq[t];
            sGt[t] = ag; sEq[t] = ae;
            ag += g1; ae += e1;
        }
    }
    __syncthreads();
    unsigned gtRun = sGt[tid], eqRun = sEq[tid];
    for (int j = 0; j < CH; ++j) {
        const int i = base + j;
        const float f = row[i];
        const unsigned k = fkey(f);
        const bool gt = k > T;
        const bool eq = (k == T);
        const bool sel = gt || (eq && eqRun < tieNeed);
        if (sel) {
            const unsigned pos = gtRun + (eqRun < tieNeed ? eqRun : tieNeed);
            tIdx[b * KSEL + pos] = i;
            tVal[b * KSEL + pos] = f;
        }
        row[i] = sel ? f : 0.0f;
        gtRun += gt; eqRun += eq;
    }
    __syncthreads();
    for (int i = tid; i < D_SAE; i += 256) g[i] = row[i];
}

// ---------------------------------------------------------------------------
// Kernel 4: sparse decode from transposed decoder (coalesced, L2-resident).
// ---------------------------------------------------------------------------
__global__ __launch_bounds__(256) void sparse_decode(const int* __restrict__ tIdx,
                                                     const float* __restrict__ tVal,
                                                     const float* __restrict__ WdT,
                                                     float* __restrict__ out) {
    __shared__ int   si[KSEL];
    __shared__ float sv[KSEL];
    const int tid = threadIdx.x;
    const size_t b = blockIdx.x;
    if (tid < KSEL) { si[tid] = tIdx[b * KSEL + tid]; sv[tid] = tVal[b * KSEL + tid]; }
    __syncthreads();
    for (int d = tid; d < D_MODEL; d += 256) {
        float acc = 0.f;
        #pragma unroll
        for (int k = 0; k < KSEL; ++k)
            acc += sv[k] * WdT[(size_t)si[k] * D_MODEL + d];
        out[b * D_MODEL + d] = acc;
    }
}

// ---------------------------------------------------------------------------
extern "C" void kernel_launch(void* const* d_in, const int* in_sizes, int n_in,
                              void* d_out, int out_size, void* d_ws, size_t ws_size,
                              hipStream_t stream) {
    (void)in_sizes; (void)n_in; (void)out_size; (void)ws_size;

    const float* X     = (const float*)d_in[0];   // [4096, 768]
    const float* W_enc = (const float*)d_in[1];   // [24576, 768]
    const float* b_enc = (const float*)d_in[2];   // [24576]
    const float* W_dec = (const float*)d_in[3];   // [768, 24576]

    float* pred    = (float*)d_out;
    float* latents = pred + (size_t)BATCH * D_MODEL;

    char* ws = (char*)d_ws;
    float*          WdT = (float*)ws;                 ws += (size_t)D_SAE * D_MODEL * sizeof(float);
    unsigned short* Wbf = (unsigned short*)ws;        ws += (size_t)D_SAE * D_MODEL * sizeof(unsigned short);
    unsigned short* Xbf = (unsigned short*)ws;        ws += (size_t)BATCH * D_MODEL * sizeof(unsigned short);
    int*            tIdx = (int*)ws;                  ws += (size_t)BATCH * KSEL * sizeof(int);
    float*          tVal = (float*)ws;

    const int nW4 = D_SAE * D_MODEL / 4;
    const int nX4 = BATCH * D_MODEL / 4;
    f32_to_bf16<<<(nX4 + 255) / 256, 256, 0, stream>>>(X, Xbf, nX4);
    f32_to_bf16<<<(nW4 + 255) / 256, 256, 0, stream>>>(W_enc, Wbf, nW4);
    wdec_transpose<<<dim3(D_SAE / 32, D_MODEL / 32), 256, 0, stream>>>(W_dec, WdT);
    encode_gemm<<<dim3(BATCH / 128, D_SAE / 128), 256, 0, stream>>>(Xbf, Wbf, b_enc, latents);
    topk_select<<<BATCH, 256, D_SAE * sizeof(float), stream>>>(latents, tIdx, tVal);
    sparse_decode<<<BATCH, 256, 0, stream>>>(tIdx, tVal, WdT, pred);
}
